// AttentionMIL_5995774345772
// MI455X (gfx1250) — compile-verified, matrix operand order checked by reading, not run on hardware
//
#include <hip/hip_runtime.h>
#include <hip/hip_bf16.h>

typedef __attribute__((ext_vector_type(16))) __bf16 v16bf;
typedef __attribute__((ext_vector_type(8)))  __bf16 v8bf;
typedef __attribute__((ext_vector_type(8)))  float  v8f;
typedef __attribute__((ext_vector_type(4)))  unsigned int u32x4;
typedef __attribute__((ext_vector_type(8)))  int i32x8;
typedef __attribute__((ext_vector_type(4)))  int i32x4;

#define B_      8
#define N_      50000
#define D_      256
#define H_      128
#define TAU_    0.3f
#define EPS_    1e-5f
#define TILE_N  128
#define NCHUNK  ((N_ + TILE_N - 1) / TILE_N)   /* 391 */
#define PSTRIDE 132
#define HPAD    136

union V16U { v16bf v; v8bf h[2]; };

__device__ __forceinline__ v8bf cvt8(float4 a, float4 b) {
    v8bf r;
    r[0]=(__bf16)a.x; r[1]=(__bf16)a.y; r[2]=(__bf16)a.z; r[3]=(__bf16)a.w;
    r[4]=(__bf16)b.x; r[5]=(__bf16)b.y; r[6]=(__bf16)b.z; r[7]=(__bf16)b.w;
    return r;
}

__device__ __forceinline__ float hsum16(float v) {
    v += __shfl_xor(v, 1); v += __shfl_xor(v, 2);
    v += __shfl_xor(v, 4); v += __shfl_xor(v, 8);
    return v;
}

// ---------------------------------------------------------------------------
// TDM: async 1-D copy of nDwords DWORDs global -> LDS via tensor descriptor.
// D# group0: count=1 | lds_addr | global_addr | type=2 ("image").
// D# group1: data_size=4B; tensor_dim0 = tile_dim0 = nDwords; dim1 = 1.
// ---------------------------------------------------------------------------
__device__ __forceinline__ void tdm_load_1d(const void* gsrc, unsigned ldsOff,
                                            unsigned nDwords) {
    unsigned long long ga = (unsigned long long)(uintptr_t)gsrc;
    u32x4 g0;
    g0[0] = 1u;                                       // count=1, user mode
    g0[1] = ldsOff;                                   // lds_addr (bytes)
    g0[2] = (unsigned)ga;                             // global_addr[31:0]
    g0[3] = (unsigned)((ga >> 32) & 0x01FFFFFFu) | 0x80000000u; // addr[56:32] | type=2
    i32x8 g1;
    g1[0] = 0x00020000;                               // data_size = 2 (4 bytes)
    g1[1] = (int)(nDwords << 16);                     // tensor_dim0[15:0]
    g1[2] = 0x00010000;                               // tensor_dim0[31:16]=0, tensor_dim1=1
    g1[3] = (int)(nDwords << 16);                     // tile_dim0
    g1[4] = 1;                                        // tile_dim1=1, tile_dim2=0
    g1[5] = (int)nDwords;                             // tensor_dim0_stride[31:0]
    g1[6] = 0;                                        // stride hi / dim1_stride lo
    g1[7] = 0;
    i32x4 gz = {0, 0, 0, 0};
#if defined(__clang_major__) && (__clang_major__ >= 23)
    i32x8 gz8 = {0, 0, 0, 0, 0, 0, 0, 0};
    __builtin_amdgcn_tensor_load_to_lds(g0, g1, gz, gz, gz8, 0);
#else
    __builtin_amdgcn_tensor_load_to_lds(g0, g1, gz, gz, 0);
#endif
}

// ---------------------------------------------------------------------------
// Kernel 0: pack W1/W2/Wa1 (fp32 [K][128]) into bf16 WMMA B-fragments.
// ---------------------------------------------------------------------------
__global__ void pack_weights(const float* __restrict__ W1,
                             const float* __restrict__ W2,
                             const float* __restrict__ Wa1,
                             __bf16* __restrict__ out) {
    int t = blockIdx.x * 256 + threadIdx.x;           // 0 .. 65535
    const float* src;
    __bf16* dst;
    int lt;
    if (t < 32768)       { src = W1;  dst = out;          lt = t; }
    else if (t < 49152)  { src = W2;  dst = out + 32768;  lt = t - 32768; }
    else                 { src = Wa1; dst = out + 49152;  lt = t - 49152; }
    int h    = lt & 15;
    int lane = (lt >> 4) & 31;
    int nt   = (lt >> 9) & 7;
    int kt   = lt >> 12;
    int hl   = lane >> 4;
    int kin  = (h & 7) + ((h >> 3) << 4) + (hl << 3);
    int K    = kt * 32 + kin;
    int col  = nt * 16 + (lane & 15);
    dst[lt] = (__bf16)src[K * 128 + col];
}

// ---------------------------------------------------------------------------
// Main fused kernel: one block = 128 instances of one bag; 8 waves x 16 rows.
// ---------------------------------------------------------------------------
__global__ __launch_bounds__(256)
void mil_main(const float* __restrict__ x,
              const __bf16* __restrict__ wfrag,
              const float* __restrict__ b1,  const float* __restrict__ g1,  const float* __restrict__ be1,
              const float* __restrict__ b2,  const float* __restrict__ g2,  const float* __restrict__ be2,
              const float* __restrict__ ba1, const float* __restrict__ Wa2, const float* __restrict__ ba2,
              float* __restrict__ partials) {
    __shared__ __align__(16) __bf16 ldsW1[32768];     // 64KB: W1f, then Wa1f
    __shared__ __align__(16) __bf16 ldsW2[16384];     // 32KB: W2f
    __shared__ __align__(16) __bf16 hbuf[TILE_N * HPAD];
    __shared__ float vec[8 * 128 + 4];
    __shared__ float sArr[TILE_N], eArr[TILE_N], Vblk[128];
    __shared__ float zAcc, sMax;

    const int tid  = threadIdx.x;
    const int lane = tid & 31;
    const int w    = tid >> 5;
    const int l15  = lane & 15;
    const int hl   = lane >> 4;
    const int bag  = blockIdx.y;
    const int chunk = blockIdx.x;

    // Kick off async TDM staging of W1f (64KB) + W2f (32KB) from wave 0.
    if (w == 0) {
        tdm_load_1d(wfrag,         (unsigned)(uintptr_t)&ldsW1[0], 16384);
        tdm_load_1d(wfrag + 32768, (unsigned)(uintptr_t)&ldsW2[0], 8192);
    }
    if (tid < 128) {
        vec[0*128+tid]=b1[tid];  vec[1*128+tid]=g1[tid];  vec[2*128+tid]=be1[tid];
        vec[3*128+tid]=b2[tid];  vec[4*128+tid]=g2[tid];  vec[5*128+tid]=be2[tid];
        vec[6*128+tid]=ba1[tid]; vec[7*128+tid]=Wa2[tid];
    }
    if (tid == 128) vec[8*128] = ba2[0];
    if (w == 0) __builtin_amdgcn_s_wait_tensorcnt(0);
    __syncthreads();

    v8f acc[8];
    #pragma unroll
    for (int nt = 0; nt < 8; ++nt)
        #pragma unroll
        for (int i = 0; i < 8; ++i) acc[nt][i] = 0.f;

    // ---------------- GEMM1: x[rows,256] @ W1 -> [rows,128] ----------------
    {
        int r0 = chunk * TILE_N + w * 16 + l15;
        if (r0 >= N_) r0 = N_ - 1;                    // clamp (masked later via e=0)
        const float* xr = x + ((size_t)bag * N_ + r0) * D_ + hl * 8;
        #pragma unroll
        for (int kt = 0; kt < 8; ++kt) {
            const float4* p = (const float4*)(xr + kt * 32);
            V16U A; A.h[0] = cvt8(p[0], p[1]); A.h[1] = cvt8(p[4], p[5]);
            #pragma unroll
            for (int g = 0; g < 2; ++g) {
                V16U Bf[4];
                #pragma unroll
                for (int q = 0; q < 4; ++q) {
                    const __bf16* wp = &ldsW1[((kt * 8 + g * 4 + q) * 32 + lane) * 16];
                    Bf[q].h[0] = *(const v8bf*)wp; Bf[q].h[1] = *(const v8bf*)(wp + 8);
                }
                #pragma unroll
                for (int q = 0; q < 4; ++q)
                    acc[g*4+q] = __builtin_amdgcn_wmma_f32_16x16x32_bf16(
                        false, A.v, false, Bf[q].v, (short)0, acc[g*4+q], false, false);
            }
        }
    }

    // LN1 + ReLU in registers; h1 -> hbuf (bf16)
    {
        float cb[8], cg[8], ce[8];
        #pragma unroll
        for (int nt = 0; nt < 8; ++nt) {
            int c = nt * 16 + l15;
            cb[nt] = vec[c]; cg[nt] = vec[128 + c]; ce[nt] = vec[256 + c];
        }
        #pragma unroll
        for (int r = 0; r < 8; ++r) {
            float s = 0.f, q = 0.f;
            #pragma unroll
            for (int nt = 0; nt < 8; ++nt) {
                float v = acc[nt][r] + cb[nt]; acc[nt][r] = v; s += v; q += v * v;
            }
            s = hsum16(s); q = hsum16(q);
            float mu = s * (1.f / 128.f);
            float rs = rsqrtf(q * (1.f / 128.f) - mu * mu + EPS_);
            int row = w * 16 + 8 * hl + r;
            #pragma unroll
            for (int nt = 0; nt < 8; ++nt) {
                float v = fmaxf((acc[nt][r] - mu) * rs * cg[nt] + ce[nt], 0.f);
                hbuf[row * HPAD + nt * 16 + l15] = (__bf16)v;
            }
        }
    }
    __syncthreads();                                  // all waves done with ldsW1 (W1f)

    // Stream Wa1f into ldsW1 asynchronously while GEMM2 runs on ldsW2.
    if (w == 0) tdm_load_1d(wfrag + 49152, (unsigned)(uintptr_t)&ldsW1[0], 8192);

    // ---------------- GEMM2: h1 @ W2 -> h2 ----------------
    #pragma unroll
    for (int nt = 0; nt < 8; ++nt)
        #pragma unroll
        for (int i = 0; i < 8; ++i) acc[nt][i] = 0.f;
    {
        const __bf16* hr = &hbuf[(w * 16 + l15) * HPAD + hl * 8];
        #pragma unroll
        for (int kt = 0; kt < 4; ++kt) {
            V16U A; A.h[0] = *(const v8bf*)(hr + kt * 32);
                    A.h[1] = *(const v8bf*)(hr + kt * 32 + 16);
            #pragma unroll
            for (int g = 0; g < 2; ++g) {
                V16U Bf[4];
                #pragma unroll
                for (int q = 0; q < 4; ++q) {
                    const __bf16* wp = &ldsW2[((kt * 8 + g * 4 + q) * 32 + lane) * 16];
                    Bf[q].h[0] = *(const v8bf*)wp; Bf[q].h[1] = *(const v8bf*)(wp + 8);
                }
                #pragma unroll
                for (int q = 0; q < 4; ++q)
                    acc[g*4+q] = __builtin_amdgcn_wmma_f32_16x16x32_bf16(
                        false, A.v, false, Bf[q].v, (short)0, acc[g*4+q], false, false);
            }
        }
    }
    __syncthreads();                                  // h1 reads done (hbuf reusable)

    // LN2 + ReLU; h2 kept fp32 in acc, bf16 copy overwrites hbuf.
    {
        float cb[8], cg[8], ce[8];
        #pragma unroll
        for (int nt = 0; nt < 8; ++nt) {
            int c = nt * 16 + l15;
            cb[nt] = vec[384 + c]; cg[nt] = vec[512 + c]; ce[nt] = vec[640 + c];
        }
        #pragma unroll
        for (int r = 0; r < 8; ++r) {
            float s = 0.f, q = 0.f;
            #pragma unroll
            for (int nt = 0; nt < 8; ++nt) {
                float v = acc[nt][r] + cb[nt]; acc[nt][r] = v; s += v; q += v * v;
            }
            s = hsum16(s); q = hsum16(q);
            float mu = s * (1.f / 128.f);
            float rs = rsqrtf(q * (1.f / 128.f) - mu * mu + EPS_);
            int row = w * 16 + 8 * hl + r;
            #pragma unroll
            for (int nt = 0; nt < 8; ++nt) {
                float v = fmaxf((acc[nt][r] - mu) * rs * cg[nt] + ce[nt], 0.f);
                acc[nt][r] = v;
                hbuf[row * HPAD + nt * 16 + l15] = (__bf16)v;
            }
        }
    }
    if (w == 0) __builtin_amdgcn_s_wait_tensorcnt(0); // Wa1f landed in ldsW1
    __syncthreads();

    // ---------------- GEMM3: h2 @ Wa1, tanh, dot Wa2 -> scores ----------------
    v8f acc3[8];
    #pragma unroll
    for (int nt = 0; nt < 8; ++nt)
        #pragma unroll
        for (int i = 0; i < 8; ++i) acc3[nt][i] = 0.f;
    {
        const __bf16* hr = &hbuf[(w * 16 + l15) * HPAD + hl * 8];
        #pragma unroll
        for (int kt = 0; kt < 4; ++kt) {
            V16U A; A.h[0] = *(const v8bf*)(hr + kt * 32);
                    A.h[1] = *(const v8bf*)(hr + kt * 32 + 16);
            #pragma unroll
            for (int g = 0; g < 2; ++g) {
                V16U Bf[4];
                #pragma unroll
                for (int q = 0; q < 4; ++q) {
                    const __bf16* wp = &ldsW1[((kt * 8 + g * 4 + q) * 32 + lane) * 16];
                    Bf[q].h[0] = *(const v8bf*)wp; Bf[q].h[1] = *(const v8bf*)(wp + 8);
                }
                #pragma unroll
                for (int q = 0; q < 4; ++q)
                    acc3[g*4+q] = __builtin_amdgcn_wmma_f32_16x16x32_bf16(
                        false, A.v, false, Bf[q].v, (short)0, acc3[g*4+q], false, false);
            }
        }
    }
    {
        float ca[8], cw[8];
        #pragma unroll
        for (int nt = 0; nt < 8; ++nt) {
            int c = nt * 16 + l15;
            ca[nt] = vec[768 + c]; cw[nt] = vec[896 + c];
        }
        float ba2v = vec[8 * 128];
        #pragma unroll
        for (int r = 0; r < 8; ++r) {
            float t = 0.f;
            #pragma unroll
            for (int nt = 0; nt < 8; ++nt) t += tanhf(acc3[nt][r] + ca[nt]) * cw[nt];
            t = hsum16(t);
            if (l15 == 0) {
                int lrow = w * 16 + 8 * hl + r;
                int grow = chunk * TILE_N + lrow;
                sArr[lrow] = (grow < N_) ? (t + ba2v) * (1.f / TAU_) : -1e30f;
            }
        }
    }
    if (tid < 128) Vblk[tid] = 0.f;
    if (tid == 0)  zAcc = 0.f;
    __syncthreads();

    if (tid < 32) {
        float m = fmaxf(fmaxf(sArr[tid], sArr[tid + 32]),
                        fmaxf(sArr[tid + 64], sArr[tid + 96]));
        m = fmaxf(m, __shfl_xor(m, 1));  m = fmaxf(m, __shfl_xor(m, 2));
        m = fmaxf(m, __shfl_xor(m, 4));  m = fmaxf(m, __shfl_xor(m, 8));
        m = fmaxf(m, __shfl_xor(m, 16));
        if (tid == 0) sMax = m;
    }
    __syncthreads();
    if (tid < 128) {
        float e = __expf(sArr[tid] - sMax);
        eArr[tid] = e;
        atomicAdd(&zAcc, e);
    }
    __syncthreads();
    {
        float ev[8];
        #pragma unroll
        for (int r = 0; r < 8; ++r) ev[r] = eArr[w * 16 + 8 * hl + r];
        #pragma unroll
        for (int nt = 0; nt < 8; ++nt) {
            float vv = 0.f;
            #pragma unroll
            for (int r = 0; r < 8; ++r) vv += ev[r] * acc[nt][r];
            atomicAdd(&Vblk[nt * 16 + l15], vv);
        }
    }
    __syncthreads();
    float* P = partials + ((size_t)bag * NCHUNK + chunk) * PSTRIDE;
    if (tid == 0) { P[0] = sMax; P[1] = zAcc; }
    if (tid < 128) P[2 + tid] = Vblk[tid];
}

// ---------------------------------------------------------------------------
// Finalize: per bag, max-rebased reduction of chunk partials + tiny heads.
// ---------------------------------------------------------------------------
__global__ void mil_finalize(const float* __restrict__ partials,
                             const float* __restrict__ Wc1, const float* __restrict__ bc1,
                             const float* __restrict__ Wc2, const float* __restrict__ bc2,
                             const float* __restrict__ Ws1, const float* __restrict__ bs1,
                             const float* __restrict__ Ws2, const float* __restrict__ bs2,
                             float* __restrict__ out) {
    __shared__ float red[128], bagv[128], hc[128], hs[64];
    __shared__ float Mv, Zv;
    const int b = blockIdx.x, tid = threadIdx.x;
    const float* P = partials + (size_t)b * NCHUNK * PSTRIDE;

    float m = -3e38f;
    for (int j = tid; j < NCHUNK; j += 128) m = fmaxf(m, P[j * PSTRIDE]);
    red[tid] = m; __syncthreads();
    for (int s = 64; s > 0; s >>= 1) {
        if (tid < s) red[tid] = fmaxf(red[tid], red[tid + s]);
        __syncthreads();
    }
    if (tid == 0) Mv = red[0];
    __syncthreads();

    float z = 0.f;
    for (int j = tid; j < NCHUNK; j += 128)
        z += P[j * PSTRIDE + 1] * __expf(P[j * PSTRIDE] - Mv);
    red[tid] = z; __syncthreads();
    for (int s = 64; s > 0; s >>= 1) {
        if (tid < s) red[tid] += red[tid + s];
        __syncthreads();
    }
    if (tid == 0) Zv = red[0];
    __syncthreads();

    float v = 0.f;
    for (int j = 0; j < NCHUNK; ++j)
        v += P[j * PSTRIDE + 2 + tid] * __expf(P[j * PSTRIDE] - Mv);
    bagv[tid] = v / Zv;
    __syncthreads();

    { float a = bc1[tid]; for (int k = 0; k < 128; ++k) a += bagv[k] * Wc1[k * 128 + tid];
      hc[tid] = fmaxf(a, 0.f); }
    if (tid < 64) {
        float a = bs1[tid];
        for (int k = 0; k < 128; ++k) a += bagv[k] * Ws1[k * 64 + tid];
        hs[tid] = fmaxf(a, 0.f);
    }
    __syncthreads();
    if (tid < 2) {
        float a = bc2[tid];
        for (int j = 0; j < 128; ++j) a += hc[j] * Wc2[j * 2 + tid];
        out[b * 2 + tid] = a;
    }
    if (tid == 2) {
        float a = bs2[0];
        for (int j = 0; j < 64; ++j) a += hs[j] * Ws2[j];
        out[B_ * 2 + b] = a;
    }
}

extern "C" void kernel_launch(void* const* d_in, const int* in_sizes, int n_in,
                              void* d_out, int out_size, void* d_ws, size_t ws_size,
                              hipStream_t stream) {
    const float* x   = (const float*)d_in[0];
    const float* W1  = (const float*)d_in[1];
    const float* b1  = (const float*)d_in[2];
    const float* g1  = (const float*)d_in[3];
    const float* be1 = (const float*)d_in[4];
    const float* W2  = (const float*)d_in[5];
    const float* b2  = (const float*)d_in[6];
    const float* g2  = (const float*)d_in[7];
    const float* be2 = (const float*)d_in[8];
    const float* Wa1 = (const float*)d_in[9];
    const float* ba1 = (const float*)d_in[10];
    const float* Wa2 = (const float*)d_in[11];
    const float* ba2 = (const float*)d_in[12];
    const float* Wc1 = (const float*)d_in[13];
    const float* bc1 = (const float*)d_in[14];
    const float* Wc2 = (const float*)d_in[15];
    const float* bc2 = (const float*)d_in[16];
    const float* Ws1 = (const float*)d_in[17];
    const float* bs1 = (const float*)d_in[18];
    const float* Ws2 = (const float*)d_in[19];
    const float* bs2 = (const float*)d_in[20];

    __bf16* wfrag   = (__bf16*)d_ws;
    float* partials = (float*)((char*)d_ws + 131072);

    hipLaunchKernelGGL(pack_weights, dim3(256), dim3(256), 0, stream, W1, W2, Wa1, wfrag);
    hipLaunchKernelGGL(mil_main, dim3(NCHUNK, B_), dim3(256), 0, stream,
                       x, wfrag, b1, g1, be1, b2, g2, be2, ba1, Wa2, ba2, partials);
    hipLaunchKernelGGL(mil_finalize, dim3(B_), dim3(128), 0, stream, partials,
                       Wc1, bc1, Wc2, bc2, Ws1, bs1, Ws2, bs2, (float*)d_out);
}
